// MuHiPAReasoning_54099408060700
// MI455X (gfx1250) — matrix operand order repaired and checked by reading, not compiled, not run on hardware
//
#include <hip/hip_runtime.h>

// ---------------------------------------------------------------------------
// BLOCK bilinear fusion pipeline for MI455X (gfx1250), bf16 WMMA + f32 accum.
//   mm1 = fusion_v2(mm, mm); mm2 = fusion_v(qq, mm1); qq2 = qq * fusion_q(qq, mm2)
// ---------------------------------------------------------------------------

typedef __attribute__((ext_vector_type(16))) __bf16 v16bf;
typedef __attribute__((ext_vector_type(8)))  float  v8f;

union AFrag {
    v16bf v;
    unsigned int u[8];
};
static_assert(sizeof(AFrag) == 32, "frag size");

__device__ __forceinline__ unsigned short f2bf(float f) {
    __bf16 h = (__bf16)f;                 // native RNE convert on gfx1250
    return __builtin_bit_cast(unsigned short, h);
}

__device__ __forceinline__ unsigned int f2bf_pk(float lo, float hi) {
    return (unsigned int)f2bf(lo) | ((unsigned int)f2bf(hi) << 16);
}

// ---------------------------------------------------------------------------
// GEMM: C[M,N] = A[M,K] @ W[K,N] + bias ; if GATE: C *= gate (elementwise).
// BM=128, BN=64, BK=64; 256 threads = 8 waves (4 wave-rows x 2 wave-cols),
// each wave computes a 32x32 tile = 2x2 WMMA subtiles over 2 k-steps/stage.
// ---------------------------------------------------------------------------
#define GBM 128
#define GBN 64
#define GBK 64
#define GSA 72   // LDS k-stride for A rows (even -> 4B-aligned bf16 pairs)
#define GSB 72   // LDS k-stride for Bt cols

template <bool GATE>
__global__ __launch_bounds__(256) void gemm_bf16_wmma(
    const float* __restrict__ A, const float* __restrict__ W,
    const float* __restrict__ bias, const float* __restrict__ gate,
    float* __restrict__ C, int M, int N, int K)
{
    __shared__ __align__(16) unsigned short As[GBM * GSA]; // [row][k]
    __shared__ __align__(16) unsigned short Bs[GBN * GSB]; // [col][k] (transposed)

    const int tid  = threadIdx.x;
    const int lane = tid & 31;
    const int wave = tid >> 5;          // 0..7
    const int wm   = wave >> 1;         // wave row (0..3) -> 32 rows each
    const int wn   = wave & 1;          // wave col (0..1) -> 32 cols each
    const int l15  = lane & 15;
    const int lhi  = lane >> 4;
    const int bm0  = blockIdx.y * GBM;
    const int bn0  = blockIdx.x * GBN;

    v8f acc[2][2] = {};

    for (int k0 = 0; k0 < K; k0 += GBK) {
        // --- stage A tile (128x64 f32 -> bf16 LDS), pairwise ---
#pragma unroll
        for (int j = 0; j < 16; ++j) {
            int p  = tid + j * 256;                 // pair index 0..4095
            int r  = p >> 5;                        // row 0..127
            int cp = (p & 31) * 2;                  // even col 0..62
            const float2 f = *(const float2*)&A[(size_t)(bm0 + r) * K + (k0 + cp)];
            *(unsigned int*)&As[r * GSA + cp] = f2bf_pk(f.x, f.y);
        }
        // --- stage W tile (64x64 f32 -> bf16 LDS, transposed) ---
#pragma unroll
        for (int j = 0; j < 16; ++j) {
            int i = tid + j * 256;                  // 0..4095
            int kr = i >> 6, cc = i & 63;
            Bs[cc * GSB + kr] = f2bf(W[(size_t)(k0 + kr) * N + (bn0 + cc)]);
        }
        // prefetch next k-stage (global_prefetch_b8 on gfx1250)
        if (k0 + GBK < K) {
            __builtin_prefetch(&A[(size_t)(bm0 + (tid >> 1)) * K + (k0 + GBK) + (tid & 1) * 32], 0, 1);
            if (tid < GBK)
                __builtin_prefetch(&W[(size_t)(k0 + GBK + tid) * N + bn0], 0, 1);
        }
        __syncthreads();

        // --- compute: 2 k-steps x 2x2 subtiles ---
#pragma unroll
        for (int ks = 0; ks < 2; ++ks) {
            AFrag af[2], bf[2];
#pragma unroll
            for (int ms = 0; ms < 2; ++ms) {
                int row = wm * 32 + ms * 16 + l15;
#pragma unroll
                for (int i = 0; i < 8; ++i) {
                    int k = ks * 32 + lhi * 16 + i * 2;
                    af[ms].u[i] = *(const unsigned int*)&As[row * GSA + k];
                }
            }
#pragma unroll
            for (int ns = 0; ns < 2; ++ns) {
                int col = wn * 32 + ns * 16 + l15;
#pragma unroll
                for (int v = 0; v < 8; ++v) {
                    int k = ks * 32 + lhi * 16 + v * 2;
                    bf[ns].u[v] = *(const unsigned int*)&Bs[col * GSB + k];
                }
            }
#pragma unroll
            for (int ms = 0; ms < 2; ++ms)
#pragma unroll
                for (int ns = 0; ns < 2; ++ns)
                    acc[ms][ns] = __builtin_amdgcn_wmma_f32_16x16x32_bf16(
                        false, af[ms].v, false, bf[ns].v,
                        (short)0, acc[ms][ns], false, false);
        }
        __syncthreads();
    }

    // --- epilogue: + bias, optional compile-time gate, store ---
#pragma unroll
    for (int ms = 0; ms < 2; ++ms)
#pragma unroll
        for (int ns = 0; ns < 2; ++ns) {
            int col = bn0 + wn * 32 + ns * 16 + l15;
            float bv = bias[col];
#pragma unroll
            for (int v = 0; v < 8; ++v) {
                int row = bm0 + wm * 32 + ms * 16 + v + lhi * 8;
                size_t off = (size_t)row * N + col;
                float cv = acc[ms][ns][v] + bv;
                if (GATE) cv *= gate[off];
                C[off] = cv;
            }
        }
}

// ---------------------------------------------------------------------------
// Fused per-chunk bilinear kernel.
// For chunk c, 64-row tile: z[64,80] = sum_r (x0c@Wm0[:,r*80:+80]+bm0)
//                                    * (x1c@Wm1[:,r*80:+80]+bm1)
// then signed sqrt + per-row L2 normalize (eps 1e-12), write z (f32).
// K=80 padded to 96 (3 WMMA k-steps); 5 waves each own 16 cols of the chunk.
// ---------------------------------------------------------------------------
#define CHM 64
#define CHK 96
#define CHS 80

__global__ __launch_bounds__(160) void chunk_bilinear_fuse(
    const float* __restrict__ X0p, const float* __restrict__ X1p,
    const float* __restrict__ Wm0, const float* __restrict__ bm0,
    const float* __restrict__ Wm1, const float* __restrict__ bm1,
    float* __restrict__ Z)
{
    const int MMD = 1600, RK = 10;
    __shared__ __align__(16) unsigned short x0s[CHM * CHK];   // [row][k]
    __shared__ __align__(16) unsigned short x1s[CHM * CHK];
    __shared__ __align__(16) unsigned short w0s[CHS * CHK];   // [col][k] (transposed)
    __shared__ __align__(16) unsigned short w1s[CHS * CHK];
    __shared__ float ztile[CHM * CHS];
    __shared__ float rinv[CHM];

    const int tid  = threadIdx.x;
    const int lane = tid & 31;
    const int wave = tid >> 5;          // 0..4, owns cols wave*16..+15
    const int l15  = lane & 15;
    const int lhi  = lane >> 4;
    const int c    = blockIdx.y;        // chunk 0..19
    const int m0g  = blockIdx.x * CHM;  // batch-row base

    // zero only the K-padding region [80,96), then stage x tiles pairwise
    for (int i = tid; i < CHM * (CHK - CHS) / 2; i += 160) {
        int r = i >> 3, kp = CHS + ((i & 7) * 2);            // 8 pairs per row
        *(unsigned int*)&x0s[r * CHK + kp] = 0u;
        *(unsigned int*)&x1s[r * CHK + kp] = 0u;
    }
    for (int p = tid; p < CHM * (CHS / 2); p += 160) {       // 2560 pairs
        int r = p / (CHS / 2), kp = (p - r * (CHS / 2)) * 2;
        size_t g = (size_t)(m0g + r) * MMD + c * CHS + kp;
        const float2 f0 = *(const float2*)&X0p[g];
        const float2 f1 = *(const float2*)&X1p[g];
        *(unsigned int*)&x0s[r * CHK + kp] = f2bf_pk(f0.x, f0.y);
        *(unsigned int*)&x1s[r * CHK + kp] = f2bf_pk(f1.x, f1.y);
    }

    v8f zacc[4] = {};

    for (int r = 0; r < RK; ++r) {
        __syncthreads();   // protect w tiles from previous iteration's readers
        // stage rank-slice weights (80x80), transposed + K-padded
        for (int i = tid; i < CHS * CHK; i += 160) {
            int col = i / CHK, k = i - col * CHK;
            unsigned short v0 = 0, v1 = 0;
            if (k < CHS) {
                size_t g = (size_t)c * CHS * 800 + (size_t)k * 800 + r * CHS + col;
                v0 = f2bf(Wm0[g]);
                v1 = f2bf(Wm1[g]);
            }
            w0s[col * CHK + k] = v0;
            w1s[col * CHK + k] = v1;
        }
        __syncthreads();

        v8f m0a[4] = {}, m1a[4] = {};
#pragma unroll
        for (int ks = 0; ks < 3; ++ks) {
            AFrag b0, b1;
            int col = wave * 16 + l15;
#pragma unroll
            for (int v = 0; v < 8; ++v) {
                int k = ks * 32 + lhi * 16 + v * 2;
                b0.u[v] = *(const unsigned int*)&w0s[col * CHK + k];
                b1.u[v] = *(const unsigned int*)&w1s[col * CHK + k];
            }
#pragma unroll
            for (int ms = 0; ms < 4; ++ms) {
                AFrag a0, a1;
                int row = ms * 16 + l15;
#pragma unroll
                for (int i = 0; i < 8; ++i) {
                    int k = ks * 32 + lhi * 16 + i * 2;
                    a0.u[i] = *(const unsigned int*)&x0s[row * CHK + k];
                    a1.u[i] = *(const unsigned int*)&x1s[row * CHK + k];
                }
                m0a[ms] = __builtin_amdgcn_wmma_f32_16x16x32_bf16(
                    false, a0.v, false, b0.v, (short)0, m0a[ms], false, false);
                m1a[ms] = __builtin_amdgcn_wmma_f32_16x16x32_bf16(
                    false, a1.v, false, b1.v, (short)0, m1a[ms], false, false);
            }
        }
        float bb0 = bm0[c * 800 + r * CHS + wave * 16 + l15];
        float bb1 = bm1[c * 800 + r * CHS + wave * 16 + l15];
#pragma unroll
        for (int ms = 0; ms < 4; ++ms)
#pragma unroll
            for (int v = 0; v < 8; ++v)
                zacc[ms][v] += (m0a[ms][v] + bb0) * (m1a[ms][v] + bb1);
    }

    __syncthreads();
    // signed sqrt; scatter accumulators (C-layout) into ztile
#pragma unroll
    for (int ms = 0; ms < 4; ++ms)
#pragma unroll
        for (int v = 0; v < 8; ++v) {
            int row = ms * 16 + v + lhi * 8;
            int col = wave * 16 + l15;
            float z = zacc[ms][v];
            float s = (z >= 0.0f) ? sqrtf(z) : -sqrtf(-z);
            ztile[row * CHS + col] = s;
        }
    __syncthreads();
    if (tid < CHM) {
        float ss = 0.0f;
#pragma unroll 8
        for (int k = 0; k < CHS; ++k) {
            float t = ztile[tid * CHS + k];
            ss += t * t;
        }
        rinv[tid] = 1.0f / fmaxf(sqrtf(ss), 1e-12f);
    }
    __syncthreads();
    for (int i = tid; i < CHM * CHS; i += 160) {
        int r = i / CHS, k = i - r * CHS;
        Z[(size_t)(m0g + r) * MMD + c * CHS + k] = ztile[i] * rinv[r];
    }
}

// ---------------------------------------------------------------------------
// Host side
// ---------------------------------------------------------------------------
struct BlockParams {
    const float *W0, *b0, *W1, *b1, *Wm0, *bm0, *Wm1, *bm1, *Wout, *bout;
};

static BlockParams load_params(void* const* d_in, int base, bool sorted_order) {
    BlockParams p;
    if (!sorted_order) {   // dict insertion order: W0,b0,W1,b1,Wm0,bm0,Wm1,bm1,Wout,bout
        p.W0   = (const float*)d_in[base + 0];
        p.b0   = (const float*)d_in[base + 1];
        p.W1   = (const float*)d_in[base + 2];
        p.b1   = (const float*)d_in[base + 3];
        p.Wm0  = (const float*)d_in[base + 4];
        p.bm0  = (const float*)d_in[base + 5];
        p.Wm1  = (const float*)d_in[base + 6];
        p.bm1  = (const float*)d_in[base + 7];
        p.Wout = (const float*)d_in[base + 8];
        p.bout = (const float*)d_in[base + 9];
    } else {               // jax pytree (alphabetical): W0,W1,Wm0,Wm1,Wout,b0,b1,bm0,bm1,bout
        p.W0   = (const float*)d_in[base + 0];
        p.W1   = (const float*)d_in[base + 1];
        p.Wm0  = (const float*)d_in[base + 2];
        p.Wm1  = (const float*)d_in[base + 3];
        p.Wout = (const float*)d_in[base + 4];
        p.b0   = (const float*)d_in[base + 5];
        p.b1   = (const float*)d_in[base + 6];
        p.bm0  = (const float*)d_in[base + 7];
        p.bm1  = (const float*)d_in[base + 8];
        p.bout = (const float*)d_in[base + 9];
    }
    return p;
}

static void run_fusion(hipStream_t s, int B,
                       const float* x0, int d0, const float* x1, int d1,
                       const BlockParams& P, int outDim, float* out,
                       const float* gate, float* x0p, float* x1p, float* zb)
{
    const int MMD = 1600;
    dim3 blk(256);
    gemm_bf16_wmma<false><<<dim3(MMD / GBN, B / GBM), blk, 0, s>>>(
        x0, P.W0, P.b0, nullptr, x0p, B, MMD, d0);
    gemm_bf16_wmma<false><<<dim3(MMD / GBN, B / GBM), blk, 0, s>>>(
        x1, P.W1, P.b1, nullptr, x1p, B, MMD, d1);
    chunk_bilinear_fuse<<<dim3(B / CHM, 20), dim3(160), 0, s>>>(
        x0p, x1p, P.Wm0, P.bm0, P.Wm1, P.bm1, zb);
    if (gate)
        gemm_bf16_wmma<true><<<dim3(outDim / GBN, B / GBM), blk, 0, s>>>(
            zb, P.Wout, P.bout, gate, out, B, outDim, MMD);
    else
        gemm_bf16_wmma<false><<<dim3(outDim / GBN, B / GBM), blk, 0, s>>>(
            zb, P.Wout, P.bout, nullptr, out, B, outDim, MMD);
}

extern "C" void kernel_launch(void* const* d_in, const int* in_sizes, int n_in,
                              void* d_out, int out_size, void* d_ws, size_t ws_size,
                              hipStream_t stream)
{
    (void)out_size; (void)ws_size;
    const int DQ = 4800, DV = 2048;
    const float* qq = (const float*)d_in[0];
    const float* mm = (const float*)d_in[1];
    const int B = in_sizes[0] / DQ;   // 4096

    // Detect flattening order of the nested param dicts:
    // insertion order puts b0 (1600) at slot 3; alphabetical puts W1 (3.2M) there.
    bool sorted_order = (n_in >= 32) && (in_sizes[3] != 1600);
    BlockParams Pv2 = load_params(d_in, 2,  sorted_order);
    BlockParams Pv  = load_params(d_in, 12, sorted_order);
    BlockParams Pq  = load_params(d_in, 22, sorted_order);

    // Workspace: x0p | x1p | z (B*1600 each) | mm1 (B*2048)
    float* ws   = (float*)d_ws;
    float* x0p  = ws;
    float* x1p  = x0p + (size_t)B * 1600;
    float* zbuf = x1p + (size_t)B * 1600;
    float* mm1  = zbuf + (size_t)B * 1600;

    float* out_mm = (float*)d_out;                    // mm_new [B, 2048]
    float* out_qq = out_mm + (size_t)B * DV;          // qq_new [B, 4800]

    // Fusion 1: mm1 = block_fusion(mm, mm, p_v2)
    run_fusion(stream, B, mm, DV, mm, DV, Pv2, DV, mm1, nullptr, x0p, x1p, zbuf);
    // Fusion 2: mm_new = block_fusion(qq, mm1, p_v) -> d_out[0]
    run_fusion(stream, B, qq, DQ, mm1, DV, Pv, DV, out_mm, nullptr, x0p, x1p, zbuf);
    // Fusion 3: qq_new = qq * block_fusion(qq, mm_new, p_q) -> d_out[1] (gate fused)
    run_fusion(stream, B, qq, DQ, out_mm, DV, Pq, DQ, out_qq, qq, x0p, x1p, zbuf);
}
